// AttentionMLP_17858474017428
// MI455X (gfx1250) — matrix operand order repaired and checked by reading, not compiled, not run on hardware
//
#include <hip/hip_runtime.h>
#include <hip/hip_bf16.h>

// ---------------------------------------------------------------------------
// AttentionMLP for MI455X (gfx1250): bf16 WMMA GEMMs with async-to-LDS B
// staging (ASYNCcnt), double-buffered pipeline, fused LN/bias/ReLU/residual.
//
// GEMM block (256 thr = 8 waves) computes a (128*MT) x 64 macro-tile; each
// wave owns MT contiguous 16-row M-tiles so every B fragment feeds MT WMMAs.
// B K-tile (32x64 bf16 = 4KB) is DMA'd global->LDS via
// global_load_async_to_lds_b128 (1 chunk/thread, XOR-swizzled vs bank
// conflicts), completion via s_wait_asynccnt + workgroup barrier.
// A fragments are two global_load_b128 per M-tile per K-step (row-major).
// All next-tile loads issue before the WMMAs of the current tile.
//
// B pack layout (convert_Bpack): dword index (kt*N + n)*16 + (k%32)/2,
// half (k&1); a lane's 8 fragment dwords are 2 contiguous 16B chunks.
// ---------------------------------------------------------------------------

typedef __attribute__((ext_vector_type(16))) __bf16 v16bf;
typedef __attribute__((ext_vector_type(8)))  float  v8f;

#define FLAG_RELU  1
#define FLAG_RES   2
#define FLAG_OBF16 4

static __device__ __forceinline__ unsigned short f32_to_bf16u(float f) {
    unsigned int u = __float_as_uint(f);
    u += 0x7FFFu + ((u >> 16) & 1u);   // round-to-nearest-even
    return (unsigned short)(u >> 16);
}

union Frag { v16bf v; uint4 q[2]; unsigned int u[8]; };

// Async DMA one 16B chunk global -> LDS (gfx1250, tracked by ASYNCcnt).
// ldsByteAddr is relative to the wave's LDS base (our __shared__ is at 0).
static __device__ __forceinline__ void async_copy_b128(unsigned ldsByteAddr,
                                                       const uint4* gsrc) {
    asm volatile("global_load_async_to_lds_b128 %0, %1, off"
                 :: "v"(ldsByteAddr), "v"(gsrc) : "memory");
}
static __device__ __forceinline__ void wait_asynccnt0() {
    asm volatile("s_wait_asynccnt 0x0" ::: "memory");
}

// ---------------------------------------------------------------------------
// C[b] = epilogue(A[b] @ Bpack + bias). Requires M%(128*MT)==0, N%64, K%32.
// ---------------------------------------------------------------------------
template <int MT>
__global__ __launch_bounds__(256)
void wmma_gemm(const unsigned short* __restrict__ A,
               const unsigned short* __restrict__ Bp,
               const float* __restrict__ bias,
               const float* __restrict__ Cres,
               float* __restrict__ Of32,
               unsigned short* __restrict__ Obf16,
               int M, int N, int K,
               long aBatchStride, long cBatchStride, int flags)
{
    __shared__ uint4 sB[2][256];                 // 2 x 4KB K-tiles of B (LDS base 0)

    const int tid   = threadIdx.x;
    const int lane  = tid & 31;
    const int lrow  = lane & 15;
    const int lhalf = lane >> 4;
    const int wid   = tid >> 5;

    const int tilesN = N >> 6;
    const int tn     = blockIdx.x % tilesN;                   // 64-wide N strip
    const int tmBase = ((blockIdx.x / tilesN) * 8 + wid) * MT;

    const long  cOff = (long)blockIdx.y * cBatchStride;
    const uint4* Ag  = (const uint4*)(A + (long)blockIdx.y * aBatchStride);
    const uint4* Bg  = (const uint4*)Bp;

    const uint4* BgT  = Bg + (long)tn * 256 + tid;            // this thread's chunk
    const long   bStep = (long)N * 4;                         // uint4 per K-tile
    const int    KT    = K >> 5;

    // swizzled LDS chunk slot (kills 4-way bank conflicts on ds_load_b128)
    const int sslot = (tid & ~3) | ((tid & 3) ^ ((tid >> 2) & 3));

    long aBase[MT];
#pragma unroll
    for (int mt = 0; mt < MT; ++mt)
        aBase[mt] = (long)((tmBase + mt) * 16 + lrow) * (K >> 3) + lhalf;

    v8f acc[MT][4];
#pragma unroll
    for (int mt = 0; mt < MT; ++mt)
#pragma unroll
        for (int s = 0; s < 4; ++s) acc[mt][s] = v8f{};

    // prologue: DMA K-tile 0 into LDS buf0; preload A tile 0 into regs
    async_copy_b128((unsigned)(sslot * 16), BgT);
    uint4 a0[MT], a1[MT], a0n[MT], a1n[MT];
#pragma unroll
    for (int mt = 0; mt < MT; ++mt) {
        a0[mt] = Ag[aBase[mt]];
        a1[mt] = Ag[aBase[mt] + 2];
        a0n[mt] = a0[mt]; a1n[mt] = a1[mt];
    }

    int buf = 0;
    for (int kt = 0; kt < KT; ++kt) {
        wait_asynccnt0();                        // own DMA chunks landed
        __syncthreads();                         // everyone's chunks landed

        if (kt + 1 < KT) {                       // issue next-tile loads early
            async_copy_b128((unsigned)(((buf ^ 1) << 12) | (sslot * 16)),
                            BgT + (long)(kt + 1) * bStep);
#pragma unroll
            for (int mt = 0; mt < MT; ++mt) {
                a0n[mt] = Ag[aBase[mt] + (long)(kt + 1) * 4];
                a1n[mt] = Ag[aBase[mt] + (long)(kt + 1) * 4 + 2];
            }
            if (kt + 2 < KT)                     // global_prefetch_b8 two ahead
                __builtin_prefetch(BgT + (long)(kt + 2) * bStep, 0, 1);
        }

        Frag a[MT];
#pragma unroll
        for (int mt = 0; mt < MT; ++mt) { a[mt].q[0] = a0[mt]; a[mt].q[1] = a1[mt]; }

        // fetch all four B fragments first -> staggered dscnt waits
        const uint4* sb4 = &sB[buf][0];
        Frag bf[4];
#pragma unroll
        for (int s = 0; s < 4; ++s) {
            const int nl = s * 16 + lrow;
            bf[s].q[0] = sb4[nl * 4 + ((lhalf)     ^ (nl & 3))];
            bf[s].q[1] = sb4[nl * 4 + ((lhalf + 2) ^ (nl & 3))];
        }
#pragma unroll
        for (int s = 0; s < 4; ++s)
#pragma unroll
            for (int mt = 0; mt < MT; ++mt)
                acc[mt][s] = __builtin_amdgcn_wmma_f32_16x16x32_bf16(
                    false, a[mt].v, false, bf[s].v, (short)0, acc[mt][s], false, false);

#pragma unroll
        for (int mt = 0; mt < MT; ++mt) { a0[mt] = a0n[mt]; a1[mt] = a1n[mt]; }
        buf ^= 1;
    }

    // epilogue: bias -> relu -> (+residual) -> store f32 or bf16
#pragma unroll
    for (int mt = 0; mt < MT; ++mt) {
#pragma unroll
        for (int s = 0; s < 4; ++s) {
            const int gc = tn * 64 + s * 16 + lrow;
            const float bv = bias[gc];
#pragma unroll
            for (int r = 0; r < 8; ++r) {
                const int  gm = (tmBase + mt) * 16 + lhalf * 8 + r;
                const long oi = cOff + (long)gm * N + gc;
                float v = acc[mt][s][r] + bv;
                if (flags & FLAG_RELU) v = fmaxf(v, 0.0f);
                if (flags & FLAG_RES)  v += Cres[oi];
                if (flags & FLAG_OBF16) Obf16[oi] = f32_to_bf16u(v);
                else                    Of32[oi]  = v;
            }
        }
    }
}

// ---------------------------------------------------------------------------
// Row LayerNorm (fp32 in) -> bf16 out, one workgroup per row.
// ---------------------------------------------------------------------------
__global__ __launch_bounds__(256)
void ln_to_bf16(const float* __restrict__ in, unsigned short* __restrict__ out,
                const float* __restrict__ g, const float* __restrict__ b, int rowLen)
{
    __shared__ float s1[256];
    __shared__ float s2[256];
    const long row = blockIdx.x;
    const float* rp = in + row * rowLen;

    float sum = 0.f, sq = 0.f;
    for (int i = threadIdx.x; i < rowLen; i += blockDim.x) {
        const float v = rp[i];
        sum += v; sq += v * v;
    }
    s1[threadIdx.x] = sum; s2[threadIdx.x] = sq;
    __syncthreads();
    for (int off = 128; off > 0; off >>= 1) {
        if ((int)threadIdx.x < off) {
            s1[threadIdx.x] += s1[threadIdx.x + off];
            s2[threadIdx.x] += s2[threadIdx.x + off];
        }
        __syncthreads();
    }
    const float mu  = s1[0] / rowLen;
    const float var = s2[0] / rowLen - mu * mu;
    const float inv = rsqrtf(var + 1e-5f);

    unsigned short* op = out + row * rowLen;
    for (int i = threadIdx.x; i < rowLen; i += blockDim.x)
        op[i] = f32_to_bf16u((rp[i] - mu) * inv * g[i] + b[i]);
}

// ---------------------------------------------------------------------------
// fp32 [K,N] -> bf16 tile-packed [Kp/32][N][16 dwords], zero-pad rows K..Kp-1.
// ---------------------------------------------------------------------------
__global__ __launch_bounds__(256)
void convert_Bpack(const float* __restrict__ in, unsigned short* __restrict__ out,
                   int K, int N, long total)
{
    const long idx = (long)blockIdx.x * blockDim.x + threadIdx.x;
    if (idx >= total) return;
    const long dw  = idx >> 1;
    const int  t   = (int)(idx & 1);
    const long kt  = dw / (16L * N);
    const long rem = dw % (16L * N);
    const int  n   = (int)(rem >> 4);
    const int  pk  = (int)(rem & 15);
    const long k   = kt * 32 + pk * 2 + t;
    const float v  = (k < K) ? in[k * N + n] : 0.0f;
    out[idx] = f32_to_bf16u(v);
}

// ---------------------------------------------------------------------------
// Build attention input t[b][k][288]: j<256 -> fc[b][j][k]; j==256 -> mean; pad.
// One wave per (b,k).
// ---------------------------------------------------------------------------
__global__ __launch_bounds__(128)
void build_t(const float* __restrict__ fc, unsigned short* __restrict__ t)
{
    const int wg   = blockIdx.x * 4 + (threadIdx.x >> 5);
    const int lane = threadIdx.x & 31;
    const int b = wg >> 7, k = wg & 127;
    const float* f = fc + (long)b * 256 * 128 + k;

    float vals[8]; float s = 0.f;
#pragma unroll
    for (int jj = 0; jj < 8; ++jj) {
        const float v = f[(long)(lane + jj * 32) * 128];
        vals[jj] = v; s += v;
    }
    for (int off = 16; off; off >>= 1) s += __shfl_xor(s, off, 32);
    const float mean = s * (1.0f / 256.0f);

    unsigned short* tr = t + ((long)b * 128 + k) * 288;
#pragma unroll
    for (int jj = 0; jj < 8; ++jj) tr[lane + jj * 32] = f32_to_bf16u(vals[jj]);
    if (lane == 0) tr[256] = f32_to_bf16u(mean);
    if (lane < 31) tr[257 + lane] = 0;
}

// ---------------------------------------------------------------------------
// out[b][k] = sum_j softmax_j(w[b][k][j]) * fc[b][j][k]. One wave per (b,k).
// ---------------------------------------------------------------------------
__global__ __launch_bounds__(128)
void softmax_dot(const float* __restrict__ w, const float* __restrict__ fc,
                 float* __restrict__ out)
{
    const int wg   = blockIdx.x * 4 + (threadIdx.x >> 5);
    const int lane = threadIdx.x & 31;
    const int b = wg >> 7, k = wg & 127;
    const float* wr = w + (long)wg * 256;

    float vals[8]; float mx = -3.4e38f;
#pragma unroll
    for (int jj = 0; jj < 8; ++jj) {
        const float v = wr[lane + jj * 32];
        vals[jj] = v; mx = fmaxf(mx, v);
    }
    for (int off = 16; off; off >>= 1) mx = fmaxf(mx, __shfl_xor(mx, off, 32));

    const float* f = fc + (long)b * 256 * 128 + k;
    float se = 0.f, num = 0.f;
#pragma unroll
    for (int jj = 0; jj < 8; ++jj) {
        const float e = __expf(vals[jj] - mx);
        se += e;
        num += e * f[(long)(lane + jj * 32) * 128];
    }
    for (int off = 16; off; off >>= 1) {
        se  += __shfl_xor(se,  off, 32);
        num += __shfl_xor(num, off, 32);
    }
    if (lane == 0) out[wg] = num / se;
}

// ---------------------------------------------------------------------------
// Host-side orchestration
// ---------------------------------------------------------------------------
extern "C" void kernel_launch(void* const* d_in, const int* in_sizes, int n_in,
                              void* d_out, int out_size, void* d_ws, size_t ws_size,
                              hipStream_t stream)
{
    (void)in_sizes; (void)n_in; (void)out_size; (void)ws_size;
    const float* x        = (const float*)d_in[0];
    const float* ln0_g    = (const float*)d_in[1];
    const float* ln0_b    = (const float*)d_in[2];
    const float* W0       = (const float*)d_in[3];
    const float* b0       = (const float*)d_in[4];
    const float* res_ln_g = (const float*)d_in[5];
    const float* res_ln_b = (const float*)d_in[6];
    const float* res_W    = (const float*)d_in[7];
    const float* res_bias = (const float*)d_in[8];
    const float* lnf_g    = (const float*)d_in[9];
    const float* lnf_b    = (const float*)d_in[10];
    const float* Wf       = (const float*)d_in[11];
    const float* bfv      = (const float*)d_in[12];
    const float* Wa1      = (const float*)d_in[13];
    const float* ba1      = (const float*)d_in[14];
    const float* Wa2      = (const float*)d_in[15];
    const float* ba2      = (const float*)d_in[16];

    const int BS = 64, LN = 256, FS = 512, H = 2048, OUT = 128, NL = 4;
    const int M = BS * LN;                       // 16384 rows
    const int A1K = 288, A1N = 3 * H;            // padded attention K, 6144

    // ---- workspace layout (bytes) ----
    char* ws = (char*)d_ws;
    const size_t off_h   = 0;                                     // f32 [M,H]
    const size_t off_hb  = off_h  + (size_t)M * H * 4;            // bf16 [M,H] (also xb, watt)
    const size_t off_W0b = off_hb + (size_t)M * H * 2;
    const size_t off_rWb = off_W0b + (size_t)FS * H * 2;
    const size_t off_Wfb = off_rWb + (size_t)NL * H * H * 2;
    const size_t off_A1b = off_Wfb + (size_t)H * OUT * 2;
    const size_t off_A2b = off_A1b + (size_t)A1K * A1N * 2;
    const size_t off_fc  = off_A2b + (size_t)A1N * LN * 2;        // f32 [M,128]
    const size_t off_t   = off_fc  + (size_t)M * OUT * 4;         // bf16 [64,128,288]
    const size_t off_hid = off_h;                                 // bf16 [64,128,6144] reuses h
    const size_t off_wat = off_hb;                                // f32 [64,128,256]   reuses hb

    float*          h    = (float*)(ws + off_h);
    unsigned short* hb   = (unsigned short*)(ws + off_hb);
    unsigned short* W0b  = (unsigned short*)(ws + off_W0b);
    unsigned short* rWb  = (unsigned short*)(ws + off_rWb);
    unsigned short* Wfb  = (unsigned short*)(ws + off_Wfb);
    unsigned short* A1b  = (unsigned short*)(ws + off_A1b);
    unsigned short* A2b  = (unsigned short*)(ws + off_A2b);
    float*          fc   = (float*)(ws + off_fc);
    unsigned short* tmat = (unsigned short*)(ws + off_t);
    unsigned short* hid  = (unsigned short*)(ws + off_hid);
    float*          watt = (float*)(ws + off_wat);

    auto cdiv = [](long a, long b) { return (int)((a + b - 1) / b); };

    // ---- weight packing fp32 -> bf16 tile-packed ----
    {
        long t0 = (long)FS * H;
        convert_Bpack<<<cdiv(t0, 256), 256, 0, stream>>>(W0, W0b, FS, H, t0);
        long t1 = (long)H * H;
        for (int i = 0; i < NL; ++i)
            convert_Bpack<<<cdiv(t1, 256), 256, 0, stream>>>(
                res_W + (long)i * H * H, rWb + (long)i * H * H, H, H, t1);
        long t2 = (long)H * OUT;
        convert_Bpack<<<cdiv(t2, 256), 256, 0, stream>>>(Wf, Wfb, H, OUT, t2);
        long t3 = (long)A1K * A1N;   // pads rows 257..287 with zeros
        convert_Bpack<<<cdiv(t3, 256), 256, 0, stream>>>(Wa1, A1b, LN + 1, A1N, t3);
        long t4 = (long)A1N * LN;
        convert_Bpack<<<cdiv(t4, 256), 256, 0, stream>>>(Wa2, A2b, A1N, LN, t4);
    }

    // ---- trunk (MT=2: block macro-tile 256x64) ----
    ln_to_bf16<<<M, 256, 0, stream>>>(x, hb, ln0_g, ln0_b, FS);
    {   // h = relu(LN(x) @ W0 + b0)   M=16384 N=2048 K=512
        wmma_gemm<2><<<dim3((M / 256) * (H / 64), 1), 256, 0, stream>>>(
            hb, W0b, b0, nullptr, h, nullptr, M, H, FS, 0, 0, FLAG_RELU);
    }
    for (int i = 0; i < NL; ++i) {
        ln_to_bf16<<<M, 256, 0, stream>>>(h, hb, res_ln_g + (long)i * H,
                                          res_ln_b + (long)i * H, H);
        wmma_gemm<2><<<dim3((M / 256) * (H / 64), 1), 256, 0, stream>>>(
            hb, rWb + (long)i * H * H, res_bias + (long)i * H,
            h, h, nullptr, M, H, H, 0, 0, FLAG_RELU | FLAG_RES);
    }
    ln_to_bf16<<<M, 256, 0, stream>>>(h, hb, lnf_g, lnf_b, H);
    {   // fc = LN(h) @ Wf + bf   M=16384 N=128 K=2048
        wmma_gemm<2><<<dim3((M / 256) * (OUT / 64), 1), 256, 0, stream>>>(
            hb, Wfb, bfv, nullptr, fc, nullptr, M, OUT, H, 0, 0, 0);
    }

    // ---- attention (MT=1: M=128) ----
    build_t<<<(BS * OUT) / 4, 128, 0, stream>>>(fc, tmat);
    {   // hid = relu(t @ Wa1 + ba1)   per-batch M=128 N=6144 K=288
        wmma_gemm<1><<<dim3((OUT / 128) * (A1N / 64), BS), 256, 0, stream>>>(
            tmat, A1b, ba1, nullptr, nullptr, hid, OUT, A1N, A1K,
            (long)OUT * A1K, (long)OUT * A1N, FLAG_RELU | FLAG_OBF16);
    }
    {   // watt = hid @ Wa2 + ba2      per-batch M=128 N=256 K=6144
        wmma_gemm<1><<<dim3((OUT / 128) * (LN / 64), BS), 256, 0, stream>>>(
            hid, A2b, ba2, nullptr, watt, nullptr, OUT, LN, A1N,
            (long)OUT * A1N, (long)OUT * LN, 0);
    }
    softmax_dot<<<(BS * OUT) / 4, 128, 0, stream>>>(watt, fc, (float*)d_out);
}
